// KGCapsuleOptimized_11063835755143
// MI455X (gfx1250) — matrix operand order repaired and checked by reading, not compiled, not run on hardware
//
#include <hip/hip_runtime.h>
#include <hip/hip_bf16.h>
#include <math.h>

// ---------------------------------------------------------------------------
// Problem constants (match reference): B=512, N=8192, D=4096, K=8, H=2048
// ---------------------------------------------------------------------------
#define CB 512
#define CN 8192
#define CD 4096
#define CK 8
#define CH 2048

typedef __bf16 bf16_t;
typedef bf16_t bf16x4  __attribute__((ext_vector_type(4)));
typedef bf16_t bf16x8  __attribute__((ext_vector_type(8)));
typedef bf16_t bf16x16 __attribute__((ext_vector_type(16)));
typedef float  floatx8 __attribute__((ext_vector_type(8)));

// Epilogue selector
#define EPI_SCALE     0   // C = alpha * acc
#define EPI_BIAS      1   // C = acc + bias[col]
#define EPI_BIAS_GELU 2   // C = gelu_exact(acc + bias[col])
#define EPI_ADDGATHER 3   // C = addsrc[add_rows[row]*add_ld + col] + alpha*acc

// ---------------------------------------------------------------------------
// bf16 WMMA GEMM: C[M,N] = epilogue( A[M,K] x B )  (f32 in, bf16 compute,
// f32 accumulate). TRANSB=0: B row-major [K,N]; TRANSB=1: B is [N,K].
// GATHER_A / GATHER_B: compile-time row-gather of A rows / B N-rows.
// Tile: 128x128x64, 256 threads = 8 waves (4 along M x 2 along N),
// 2 K-steps x (2x4) fragments of v_wmma_f32_16x16x32_bf16 per iteration.
// Pipeline: regs prefetch (k0+64) + L2 prefetch (k0+128) overlap compute;
// LDS is double-buffered -> ONE barrier per K-iteration.
// All dims are multiples of the tile for this problem (no bounds checks).
// ---------------------------------------------------------------------------
template <bool TRANSB, bool GATHER_A, bool GATHER_B, int EPI>
__global__ __launch_bounds__(256) void gemm_bf16_wmma_kernel(
    int M, int N, int Kd,
    const float* __restrict__ A, int lda, const int* __restrict__ a_rows,
    const float* __restrict__ B, int ldb, const int* __restrict__ b_rows,
    float* __restrict__ C, int ldc,
    float alpha,
    const float* __restrict__ bias,
    const float* __restrict__ addsrc, int add_ld, const int* __restrict__ add_rows)
{
    constexpr int BK   = 64;  // staged K per iteration (2 WMMA k-steps)
    constexpr int LDSS = 72;  // padded LDS row stride (144B: 16B-aligned rows)

    const int tid   = threadIdx.x;
    const int lane  = tid & 31;
    const int wid   = tid >> 5;
    const int half  = (lane >> 4) & 1;
    const int l16   = lane & 15;
    const int waveM = wid & 3;   // 4 waves along M -> 32 rows each
    const int waveN = wid >> 2;  // 2 waves along N -> 64 cols each
    const int gm    = blockIdx.y * 128;
    const int gn    = blockIdx.x * 128;

    // double-buffered LDS tiles (ping-pong): 4 x 18KB = 72KB
    __shared__ __align__(16) bf16_t As0[128 * LDSS];
    __shared__ __align__(16) bf16_t As1[128 * LDSS];
    __shared__ __align__(16) bf16_t Bs0[128 * LDSS];
    __shared__ __align__(16) bf16_t Bs1[128 * LDSS];

    // ---- K-invariant staging descriptors (gather indices hoisted) ---------
    const float* aSrc[8];
    #pragma unroll
    for (int i = 0; i < 8; ++i) {
        const int idx4 = tid + i * 256;       // 0..2047 quads (128x64 / 4)
        const int off  = idx4 * 4;
        const int r    = off >> 6;            // row 0..127
        const int kk   = off & 63;
        const int srow = GATHER_A ? a_rows[gm + r] : (gm + r);
        aSrc[i] = A + (size_t)srow * lda + kk;
    }
    const float* bSrc[8];
    #pragma unroll
    for (int i = 0; i < 8; ++i) {
        const int idx4 = tid + i * 256;
        if (TRANSB) {
            const int off  = idx4 * 4;
            const int n    = off >> 6;        // n-row 0..127
            const int kk   = off & 63;
            const int srow = GATHER_B ? b_rows[gn + n] : (gn + n);
            bSrc[i] = B + (size_t)srow * ldb + kk;
        } else {
            const int n  = idx4 & 127;        // coalesced along N
            const int kk = (idx4 >> 7) * 4;   // 0..60
            bSrc[i] = B + (size_t)kk * ldb + gn + n;
        }
    }

    float4 aR[8];
    float4 bRt[8];     // TRANSB staging regs
    float  bRs[8][4];  // non-trans staging regs (K-strided scalars)

    auto load_tiles = [&](int k0) {
        #pragma unroll
        for (int i = 0; i < 8; ++i) aR[i] = *(const float4*)(aSrc[i] + k0);
        if (TRANSB) {
            #pragma unroll
            for (int i = 0; i < 8; ++i) bRt[i] = *(const float4*)(bSrc[i] + k0);
        } else {
            #pragma unroll
            for (int i = 0; i < 8; ++i)
                #pragma unroll
                for (int j = 0; j < 4; ++j)
                    bRs[i][j] = bSrc[i][(size_t)(k0 + j) * ldb];
        }
    };

    // speculative L2 prefetch of the tile after next (global_prefetch_b8;
    // does not touch LOADcnt)
    auto prefetch_tiles = [&](int k0) {
        #pragma unroll
        for (int i = 0; i < 8; ++i) __builtin_prefetch(aSrc[i] + k0, 0, 1);
        if (TRANSB) {
            #pragma unroll
            for (int i = 0; i < 8; ++i) __builtin_prefetch(bSrc[i] + k0, 0, 1);
        } else {
            #pragma unroll
            for (int i = 0; i < 8; ++i)
                __builtin_prefetch(bSrc[i] + (size_t)k0 * ldb, 0, 1);
        }
    };

    auto store_tiles = [&](bf16_t* __restrict__ Ad, bf16_t* __restrict__ Bd) {
        #pragma unroll
        for (int i = 0; i < 8; ++i) {
            const int idx4 = tid + i * 256;
            const int off  = idx4 * 4;
            bf16x4 p;
            p[0] = (bf16_t)aR[i].x; p[1] = (bf16_t)aR[i].y;
            p[2] = (bf16_t)aR[i].z; p[3] = (bf16_t)aR[i].w;
            *(bf16x4*)&Ad[(off >> 6) * LDSS + (off & 63)] = p;   // ds_store_b64
        }
        #pragma unroll
        for (int i = 0; i < 8; ++i) {
            const int idx4 = tid + i * 256;
            bf16x4 p;
            int n, kk;
            if (TRANSB) {
                const int off = idx4 * 4;
                n = off >> 6; kk = off & 63;
                p[0] = (bf16_t)bRt[i].x; p[1] = (bf16_t)bRt[i].y;
                p[2] = (bf16_t)bRt[i].z; p[3] = (bf16_t)bRt[i].w;
            } else {
                n = idx4 & 127; kk = (idx4 >> 7) * 4;
                p[0] = (bf16_t)bRs[i][0]; p[1] = (bf16_t)bRs[i][1];
                p[2] = (bf16_t)bRs[i][2]; p[3] = (bf16_t)bRs[i][3];
            }
            *(bf16x4*)&Bd[n * LDSS + kk] = p;                    // ds_store_b64
        }
    };

    floatx8 acc[2][4];
    #pragma unroll
    for (int i = 0; i < 2; ++i)
        #pragma unroll
        for (int j = 0; j < 4; ++j)
            #pragma unroll
            for (int r = 0; r < 8; ++r)
                acc[i][j][r] = 0.0f;

    auto compute = [&](const bf16_t* __restrict__ Ab, const bf16_t* __restrict__ Bb) {
        #pragma unroll
        for (int s = 0; s < 2; ++s) {           // two 16x16x32 K-steps
            // A 16x32 frag: lane m=l16; e<8 -> k=8*half+e ; e>=8 -> k=16+8*half+(e-8)
            bf16x16 afrag[2];
            #pragma unroll
            for (int f = 0; f < 2; ++f) {
                const int r = waveM * 32 + f * 16 + l16;
                const bf16x8 lo = *(const bf16x8*)&Ab[r * LDSS + s * 32 + 8 * half];
                const bf16x8 hi = *(const bf16x8*)&Ab[r * LDSS + s * 32 + 16 + 8 * half];
                afrag[f] = __builtin_shufflevector(lo, hi, 0, 1, 2, 3, 4, 5, 6, 7,
                                                   8, 9, 10, 11, 12, 13, 14, 15);
            }
            // B 32x16 frag: lane n=l16; e<8 -> k=8*half+e ; e>=8 -> k=16+8*half+(e-8)
            bf16x16 bfrag[4];
            #pragma unroll
            for (int f = 0; f < 4; ++f) {
                const int n = waveN * 64 + f * 16 + l16;
                const bf16x8 lo = *(const bf16x8*)&Bb[n * LDSS + s * 32 + 8 * half];
                const bf16x8 hi = *(const bf16x8*)&Bb[n * LDSS + s * 32 + 16 + 8 * half];
                bfrag[f] = __builtin_shufflevector(lo, hi, 0, 1, 2, 3, 4, 5, 6, 7,
                                                   8, 9, 10, 11, 12, 13, 14, 15);
            }
            #pragma unroll
            for (int i = 0; i < 2; ++i)
                #pragma unroll
                for (int j = 0; j < 4; ++j)
                    acc[i][j] = __builtin_amdgcn_wmma_f32_16x16x32_bf16(
                        false, afrag[i], false, bfrag[j],
                        (short)0, acc[i][j], false, false);
        }
    };

    // ---- pipelined main loop: one barrier per K-iteration -----------------
    const bf16_t* cA = As0; const bf16_t* cB = Bs0;   // compute buffers
    bf16_t*       nA = As1; bf16_t*       nB = Bs1;   // staging buffers

    load_tiles(0);
    store_tiles((bf16_t*)cA, (bf16_t*)cB);
    __syncthreads();

    for (int k0 = 0; k0 < Kd; k0 += BK) {
        const bool more = (k0 + BK < Kd);
        if (more) {
            load_tiles(k0 + BK);                      // HBM -> regs (overlaps WMMA)
            if (k0 + 2 * BK < Kd) prefetch_tiles(k0 + 2 * BK); // -> L2
        }
        compute(cA, cB);                              // LDS -> WMMA
        if (more) {
            store_tiles(nA, nB);                      // regs -> other LDS buffer
            const bf16_t* tA = cA; cA = nA; nA = (bf16_t*)tA;
            const bf16_t* tB = cB; cB = nB; nB = (bf16_t*)tB;
            __syncthreads();
        }
    }

    // ---- epilogue: C/D layout -> M = r + 8*half, N = l16 ------------------
    #pragma unroll
    for (int i = 0; i < 2; ++i) {
        #pragma unroll
        for (int j = 0; j < 4; ++j) {
            #pragma unroll
            for (int r = 0; r < 8; ++r) {
                const int row = gm + waveM * 32 + i * 16 + r + 8 * half;
                const int col = gn + waveN * 64 + j * 16 + l16;
                float v = acc[i][j][r];
                if (EPI == EPI_SCALE) {
                    v *= alpha;
                } else if (EPI == EPI_BIAS) {
                    v += bias[col];
                } else if (EPI == EPI_BIAS_GELU) {
                    v += bias[col];
                    v = 0.5f * v * (1.0f + erff(v * 0.70710678118654752f));
                } else { // EPI_ADDGATHER
                    v = addsrc[(size_t)add_rows[row] * add_ld + col] + alpha * v;
                }
                C[(size_t)row * ldc + col] = v;
            }
        }
    }
}

// ---------------------------------------------------------------------------
// Fused softmax (max + expsum) and iterative top-K argmax per score row.
// Softmax is monotone => top-K of scores == top-K of attn.
// ---------------------------------------------------------------------------
__global__ __launch_bounds__(256) void softmax_topk_kernel(
    const float* __restrict__ scores, int N, int Kt,
    float* __restrict__ topk_w, int* __restrict__ topk_i)
{
    const int row = blockIdx.x;
    const int tid = threadIdx.x;
    const float* s = scores + (size_t)row * N;

    __shared__ float rv[256];
    __shared__ int   ri[256];
    __shared__ int   chosen[16];

    float m = -INFINITY;
    for (int i = tid; i < N; i += 256) m = fmaxf(m, s[i]);
    rv[tid] = m; __syncthreads();
    for (int off = 128; off > 0; off >>= 1) {
        if (tid < off) rv[tid] = fmaxf(rv[tid], rv[tid + off]);
        __syncthreads();
    }
    const float mx = rv[0]; __syncthreads();

    float sum = 0.0f;
    for (int i = tid; i < N; i += 256) sum += expf(s[i] - mx);
    rv[tid] = sum; __syncthreads();
    for (int off = 128; off > 0; off >>= 1) {
        if (tid < off) rv[tid] += rv[tid + off];
        __syncthreads();
    }
    const float tot = rv[0]; __syncthreads();

    for (int k = 0; k < Kt; ++k) {
        float bv = -INFINITY; int bi = 0x7fffffff;
        for (int i = tid; i < N; i += 256) {
            bool used = false;
            for (int c = 0; c < k; ++c) used |= (chosen[c] == i);
            if (used) continue;
            const float v = s[i];
            if (v > bv || (v == bv && i < bi)) { bv = v; bi = i; }
        }
        rv[tid] = bv; ri[tid] = bi; __syncthreads();
        for (int off = 128; off > 0; off >>= 1) {
            if (tid < off) {
                if (rv[tid + off] > rv[tid] ||
                    (rv[tid + off] == rv[tid] && ri[tid + off] < ri[tid])) {
                    rv[tid] = rv[tid + off]; ri[tid] = ri[tid + off];
                }
            }
            __syncthreads();
        }
        if (tid == 0) {
            chosen[k] = ri[0];
            topk_i[row * Kt + k] = ri[0];
            topk_w[row * Kt + k] = expf(rv[0] - mx) / tot;
        }
        __syncthreads();
    }
}

// ---------------------------------------------------------------------------
// xagg[b,d] = x[b,d] + sum_k h[(b*K+k),d] * w[b*K+k]
// ---------------------------------------------------------------------------
__global__ __launch_bounds__(256) void agg_residual_kernel(
    const float* __restrict__ h, const float* __restrict__ w,
    const float* __restrict__ x, float* __restrict__ xagg, int Bn, int Dn, int Kt)
{
    const int idx = blockIdx.x * blockDim.x + threadIdx.x;
    if (idx >= Bn * Dn) return;
    const int b = idx / Dn;
    const int d = idx - b * Dn;
    float a = 0.0f;
    #pragma unroll
    for (int k = 0; k < CK; ++k)
        a += h[(size_t)(b * Kt + k) * Dn + d] * w[b * Kt + k];
    xagg[idx] = x[idx] + a;
}

// ---------------------------------------------------------------------------
// conf[row] = sigmoid( dot(final[row,:], Wc) )
// ---------------------------------------------------------------------------
__global__ __launch_bounds__(256) void conf_kernel(
    const float* __restrict__ finalv, const float* __restrict__ Wc,
    float* __restrict__ conf, int Dn)
{
    const int row = blockIdx.x;
    __shared__ float red[256];
    float sum = 0.0f;
    for (int i = threadIdx.x; i < Dn; i += 256)
        sum += finalv[(size_t)row * Dn + i] * Wc[i];
    red[threadIdx.x] = sum; __syncthreads();
    for (int off = 128; off > 0; off >>= 1) {
        if (threadIdx.x < off) red[threadIdx.x] += red[threadIdx.x + off];
        __syncthreads();
    }
    if (threadIdx.x == 0) conf[row] = 1.0f / (1.0f + expf(-red[0]));
}

// ---------------------------------------------------------------------------
// Launch: full pipeline.
// ---------------------------------------------------------------------------
extern "C" void kernel_launch(void* const* d_in, const int* in_sizes, int n_in,
                              void* d_out, int out_size, void* d_ws, size_t ws_size,
                              hipStream_t stream) {
    const float* x        = (const float*)d_in[0];   // [B,D]
    const float* node_emb = (const float*)d_in[1];   // [N,D]
    const float* adj      = (const float*)d_in[2];   // [N,N]
    const float* Wq       = (const float*)d_in[3];   // [D,D]
    const float* Wk       = (const float*)d_in[4];   // [D,D]
    const float* W1       = (const float*)d_in[5];   // [D,H]
    const float* b1       = (const float*)d_in[6];   // [H]
    const float* W2       = (const float*)d_in[7];   // [H,D]
    const float* b2       = (const float*)d_in[8];   // [D]
    const float* Wc       = (const float*)d_in[9];   // [D,1]
    const float* Wo       = (const float*)d_in[10];  // [D,D]
    const float* bo       = (const float*)d_in[11];  // [D]

    const int Bn = CB, Nn = CN, Dn = CD, Kt = CK, Hn = CH;
    const int BK = Bn * Kt; // 4096 bundle rows

    // workspace layout (256B aligned slabs; later stages alias dead buffers)
    char* ws = (char*)d_ws;
    size_t off = 0;
    auto alloc = [&](size_t bytes) -> void* {
        void* p = ws + off;
        off += (bytes + 255) & ~(size_t)255;
        return p;
    };
    float* q      = (float*)alloc((size_t)Bn * Dn * 4);   // dead after scores
    float* keys   = (float*)alloc((size_t)Nn * Dn * 4);   // dead after scores
    float* scores = (float*)alloc((size_t)Bn * Nn * 4);   // dead after topk
    float* topk_w = (float*)alloc((size_t)Bn * Kt * 4);
    int*   topk_i = (int*)  alloc((size_t)Bn * Kt * 4);   // flat [B*K] gather list
    float* msg    = (float*)alloc((size_t)BK * Dn * 4);   // dead after enh
    float* enh    = (float*)alloc((size_t)BK * Dn * 4);
    float* hmid   = msg;                                  // alias into msg slab
    float* h      = keys;                                 // alias into keys slab
    float* xagg   = q;                                    // alias into q slab

    float* outF = (float*)d_out;              // final: [B,D]
    float* outC = outF + (size_t)Bn * Dn;     // conf : [B,1]

    const dim3 blk(256);
    const float inv_sqrt_d = 0.015625f; // 1/sqrt(4096)

    // 1) q = x @ Wq
    gemm_bf16_wmma_kernel<false, false, false, EPI_SCALE>
        <<<dim3(Dn / 128, Bn / 128), blk, 0, stream>>>(
            Bn, Dn, Dn, x, Dn, nullptr, Wq, Dn, nullptr, q, Dn,
            1.0f, nullptr, nullptr, 0, nullptr);

    // 2) keys = node_emb @ Wk
    gemm_bf16_wmma_kernel<false, false, false, EPI_SCALE>
        <<<dim3(Dn / 128, Nn / 128), blk, 0, stream>>>(
            Nn, Dn, Dn, node_emb, Dn, nullptr, Wk, Dn, nullptr, keys, Dn,
            1.0f, nullptr, nullptr, 0, nullptr);

    // 3) scores = (q @ keys^T) / sqrt(D)
    gemm_bf16_wmma_kernel<true, false, false, EPI_SCALE>
        <<<dim3(Nn / 128, Bn / 128), blk, 0, stream>>>(
            Bn, Nn, Dn, q, Dn, nullptr, keys, Dn, nullptr, scores, Nn,
            inv_sqrt_d, nullptr, nullptr, 0, nullptr);

    // 4) softmax + top-8 (weights + flat gather indices)
    softmax_topk_kernel<<<Bn, blk, 0, stream>>>(scores, Nn, Kt, topk_w, topk_i);

    // 5) msg = adj[topk_i] @ node_emb          (A rows gathered from adj)
    gemm_bf16_wmma_kernel<false, true, false, EPI_SCALE>
        <<<dim3(Dn / 128, BK / 128), blk, 0, stream>>>(
            BK, Dn, Nn, adj, Nn, topk_i, node_emb, Dn, nullptr, msg, Dn,
            1.0f, nullptr, nullptr, 0, nullptr);

    // 6) enh = sel + 0.1 * (msg @ sel^T)   [bundle_inf = Msg @ Sel^T from t.T]
    gemm_bf16_wmma_kernel<true, false, true, EPI_ADDGATHER>
        <<<dim3(Dn / 128, BK / 128), blk, 0, stream>>>(
            BK, Dn, Dn, msg, Dn, nullptr, node_emb, Dn, topk_i, enh, Dn,
            0.1f, nullptr, node_emb, Dn, topk_i);

    // 7) hmid = gelu_exact(enh @ W1 + b1)
    gemm_bf16_wmma_kernel<false, false, false, EPI_BIAS_GELU>
        <<<dim3(Hn / 128, BK / 128), blk, 0, stream>>>(
            BK, Hn, Dn, enh, Dn, nullptr, W1, Hn, nullptr, hmid, Hn,
            1.0f, b1, nullptr, 0, nullptr);

    // 8) h = hmid @ W2 + b2
    gemm_bf16_wmma_kernel<false, false, false, EPI_BIAS>
        <<<dim3(Dn / 128, BK / 128), blk, 0, stream>>>(
            BK, Dn, Hn, hmid, Hn, nullptr, W2, Dn, nullptr, h, Dn,
            1.0f, b2, nullptr, 0, nullptr);

    // 9) xagg = x + sum_k w[b,k] * h[b,k,:]
    agg_residual_kernel<<<(Bn * Dn + 255) / 256, blk, 0, stream>>>(
        h, topk_w, x, xagg, Bn, Dn, Kt);

    // 10) final = xagg @ Wo + bo  -> d_out
    gemm_bf16_wmma_kernel<false, false, false, EPI_BIAS>
        <<<dim3(Dn / 128, Bn / 128), blk, 0, stream>>>(
            Bn, Dn, Dn, xagg, Dn, nullptr, Wo, Dn, nullptr, outF, Dn,
            1.0f, bo, nullptr, 0, nullptr);

    // 11) conf = sigmoid(final @ Wc) -> d_out tail
    conf_kernel<<<Bn, blk, 0, stream>>>(outF, Wc, outC, Dn);
}
